// RopeBlock_16020228014147
// MI455X (gfx1250) — matrix operand order repaired and checked by reading, not compiled, not run on hardware
//
#include <hip/hip_runtime.h>
#include <cstdint>
#include <cstddef>

#define D_MODELc 1024
#define N_HEADSc 16
#define D_HEADc  64
#define D_MLPc   4096
#define BATCHc   2
#define SEQc     2048
#define NTOKc    (BATCHc*SEQc)   // 4096

typedef __attribute__((ext_vector_type(8)))  float  v8f;
typedef __attribute__((ext_vector_type(8)))  __bf16 v8bf;
typedef __attribute__((ext_vector_type(16))) __bf16 v16bf;

static __device__ __forceinline__ v16bf cat8(v8bf lo, v8bf hi) {
  return __builtin_shufflevector(lo, hi, 0,1,2,3,4,5,6,7,8,9,10,11,12,13,14,15);
}
static __device__ __forceinline__ v8f zero8() {
  v8f r;
#pragma unroll
  for (int i = 0; i < 8; ++i) r[i] = 0.0f;
  return r;
}
static __device__ __forceinline__ v8f wmma_bf16(v16bf a, v16bf b, v8f c) {
  return __builtin_amdgcn_wmma_f32_16x16x32_bf16(false, a, false, b, (short)0, c, false, false);
}

// Async global -> LDS copy (16B per lane), tracked by ASYNCcnt.
// lds_off is the wave-relative LDS byte address (low 32 bits of generic ptr).
static __device__ __forceinline__ void async_copy_b128(uint32_t lds_off, const void* g) {
  asm volatile("global_load_async_to_lds_b128 %0, %1, off"
               :: "v"(lds_off), "v"(g) : "memory");
}
static __device__ __forceinline__ uint32_t lds_addr(const void* p) {
  return (uint32_t)(uintptr_t)p;   // AS3 offset lives in the low dword
}

static __device__ __forceinline__ float fast_tanh(float y) {
  y = fminf(fmaxf(y, -15.0f), 15.0f);
  float e = __expf(2.0f * y);
  return (e - 1.0f) * __builtin_amdgcn_rcpf(e + 1.0f);
}

// ---------------------------------------------------------------------------
// Weight transpose + f32 -> bf16 cast.  in: (H, C, D) f32, out: [(h*D+d)][C] bf16
// ---------------------------------------------------------------------------
__global__ __launch_bounds__(256) void transpose_w_kernel(
    const float* __restrict__ W, __bf16* __restrict__ Wt, int H, int C, int D) {
  size_t idx = (size_t)blockIdx.x * 256 + threadIdx.x;
  int d = (int)(idx % D);
  size_t rem = idx / D;
  int c = (int)(rem % C);
  int h = (int)(rem / C);
  Wt[((size_t)(h * D + d)) * C + c] = (__bf16)W[idx];
}

// ---------------------------------------------------------------------------
// LayerNorm (no eps, matching reference) + bf16 cast.  One block per row.
// ---------------------------------------------------------------------------
__global__ __launch_bounds__(256) void ln_bf16_kernel(
    const float* __restrict__ x, const float* __restrict__ w,
    const float* __restrict__ b, __bf16* __restrict__ out) {
  __shared__ float red[256];
  const int row = blockIdx.x, tid = threadIdx.x;
  const float* xr = x + (size_t)row * D_MODELc;
  float v[4];
  float s = 0.0f;
#pragma unroll
  for (int i = 0; i < 4; ++i) { v[i] = xr[tid * 4 + i]; s += v[i]; }
  red[tid] = s;
  __syncthreads();
  for (int off = 128; off > 0; off >>= 1) {
    if (tid < off) red[tid] += red[tid + off];
    __syncthreads();
  }
  float mean = red[0] * (1.0f / D_MODELc);
  __syncthreads();
  s = 0.0f;
#pragma unroll
  for (int i = 0; i < 4; ++i) { v[i] -= mean; s += v[i] * v[i]; }
  red[tid] = s;
  __syncthreads();
  for (int off = 128; off > 0; off >>= 1) {
    if (tid < off) red[tid] += red[tid + off];
    __syncthreads();
  }
  float rstd = rsqrtf(red[0] * (1.0f / D_MODELc));
#pragma unroll
  for (int i = 0; i < 4; ++i) {
    int idx = tid * 4 + i;
    out[(size_t)row * D_MODELc + idx] = (__bf16)(v[i] * rstd * w[idx] + b[idx]);
  }
}

// ---------------------------------------------------------------------------
// Tiled WMMA bf16 GEMM with double-buffered async (ASYNCcnt) LDS staging:
//   out = act(A[M,K] * Bt[N,K]^T + bias) (+ resid)
// Block tile 128x128, 8 waves, wave tile 32x64 (2x4 WMMA), K step 32.
// M,N multiples of 128; K multiple of 32.  bias is always present.
// ---------------------------------------------------------------------------
template <int ACT, int RESID, int OUTBF>
__global__ __launch_bounds__(256) void gemm_kernel(
    const __bf16* __restrict__ A, const __bf16* __restrict__ Bt,
    const float* __restrict__ bias, const float* __restrict__ resid,
    float* __restrict__ outF, __bf16* __restrict__ outB, int N, int K) {
  __shared__ __bf16 As[2][128 * 40];
  __shared__ __bf16 Bs[2][128 * 40];
  const int tid = threadIdx.x;
  const int lane = tid & 31, wave = tid >> 5;
  const int lrow = lane & 15, lhi = lane >> 4;
  const int wr = wave >> 1, wc = wave & 1;
  const int m0 = blockIdx.y * 128, n0 = blockIdx.x * 128;
  const int kA = lhi * 8, kB = lhi * 8;
  // each thread owns two 16B chunks of each tile
  const int c0 = tid, c1 = tid + 256;
  const int r0 = c0 >> 2, q0 = (c0 & 3) * 8;
  const int r1 = c1 >> 2, q1 = (c1 & 3) * 8;

  v8f acc[2][4];
#pragma unroll
  for (int i = 0; i < 2; ++i)
#pragma unroll
    for (int j = 0; j < 4; ++j) acc[i][j] = zero8();

  // prologue: issue tile 0 into buffer 0
  {
    async_copy_b128(lds_addr(&As[0][r0 * 40 + q0]), &A[(size_t)(m0 + r0) * K + q0]);
    async_copy_b128(lds_addr(&Bs[0][r0 * 40 + q0]), &Bt[(size_t)(n0 + r0) * K + q0]);
    async_copy_b128(lds_addr(&As[0][r1 * 40 + q1]), &A[(size_t)(m0 + r1) * K + q1]);
    async_copy_b128(lds_addr(&Bs[0][r1 * 40 + q1]), &Bt[(size_t)(n0 + r1) * K + q1]);
  }

  const int nk = K >> 5;
  for (int ki = 0; ki < nk; ++ki) {
    const int buf = ki & 1;
    if (ki + 1 < nk) {
      const int k1 = (ki + 1) << 5;
      async_copy_b128(lds_addr(&As[buf ^ 1][r0 * 40 + q0]), &A[(size_t)(m0 + r0) * K + k1 + q0]);
      async_copy_b128(lds_addr(&Bs[buf ^ 1][r0 * 40 + q0]), &Bt[(size_t)(n0 + r0) * K + k1 + q0]);
      async_copy_b128(lds_addr(&As[buf ^ 1][r1 * 40 + q1]), &A[(size_t)(m0 + r1) * K + k1 + q1]);
      async_copy_b128(lds_addr(&Bs[buf ^ 1][r1 * 40 + q1]), &Bt[(size_t)(n0 + r1) * K + k1 + q1]);
      asm volatile("s_wait_asynccnt 0x4" ::: "memory");  // tile ki landed
    } else {
      asm volatile("s_wait_asynccnt 0x0" ::: "memory");
    }
    __syncthreads();  // tile ki visible to all waves
    v16bf af[2], bf[4];
#pragma unroll
    for (int i = 0; i < 2; ++i) {
      const __bf16* p = &As[buf][(wr * 32 + i * 16 + lrow) * 40 + kA];
      af[i] = cat8(*(const v8bf*)p, *(const v8bf*)(p + 16));
    }
#pragma unroll
    for (int j = 0; j < 4; ++j) {
      const __bf16* p = &Bs[buf][(wc * 64 + j * 16 + lrow) * 40 + kB];
      bf[j] = cat8(*(const v8bf*)p, *(const v8bf*)(p + 16));
    }
#pragma unroll
    for (int i = 0; i < 2; ++i)
#pragma unroll
      for (int j = 0; j < 4; ++j) acc[i][j] = wmma_bf16(af[i], bf[j], acc[i][j]);
    __syncthreads();  // all waves done with buf before it is overwritten
  }

#pragma unroll
  for (int i = 0; i < 2; ++i)
#pragma unroll
    for (int j = 0; j < 4; ++j)
#pragma unroll
      for (int r = 0; r < 8; ++r) {
        int m = m0 + wr * 32 + i * 16 + r + lhi * 8;
        int n = n0 + wc * 64 + j * 16 + lrow;
        float v = acc[i][j][r] + bias[n];
        if (ACT) {
          float t = v;
          v = 0.5f * t * (1.0f + fast_tanh(0.7978845608f * (t + 0.044715f * t * t * t)));
        }
        if (RESID) v += resid[(size_t)m * N + n];
        if (OUTBF) outB[(size_t)m * N + n] = (__bf16)v;
        else       outF[(size_t)m * N + n] = v;
      }
}

// ---------------------------------------------------------------------------
// RoPE on Q,K + transpose V.  One thread per (b, t, h, pair i in [0,32)).
// QKV: [b*T][3072] bf16 (Q | K | V blocks of 1024 cols).
// Qr/Kr: [(b*H+h)][t][64] bf16;  Vt: [(b*H+h)][64][t] bf16.
// ---------------------------------------------------------------------------
__global__ __launch_bounds__(256) void reformat_qkv_kernel(
    const __bf16* __restrict__ QKV, __bf16* __restrict__ Qr,
    __bf16* __restrict__ Kr, __bf16* __restrict__ Vt) {
  int idx = blockIdx.x * 256 + threadIdx.x;
  int i = idx & 31;
  int h = (idx >> 5) & (N_HEADSc - 1);
  int t = (idx >> 9) & (SEQc - 1);
  int b = idx >> 20;
  float inv = __expf(-((float)(2 * i) / 64.0f) * 9.210340371976184f); // ln(1e4)
  float ang = (float)t * inv;
  float sn, cs;
  __sincosf(ang, &sn, &cs);
  size_t src = ((size_t)(b * SEQc + t)) * 3072 + h * D_HEADc + 2 * i;
  size_t dst = ((size_t)(b * N_HEADSc + h) * SEQc + t) * D_HEADc + 2 * i;
  float xr = (float)QKV[src], xi = (float)QKV[src + 1];
  Qr[dst]     = (__bf16)(xr * cs - xi * sn);
  Qr[dst + 1] = (__bf16)(xr * sn + xi * cs);
  xr = (float)QKV[src + 1024]; xi = (float)QKV[src + 1025];
  Kr[dst]     = (__bf16)(xr * cs - xi * sn);
  Kr[dst + 1] = (__bf16)(xr * sn + xi * cs);
  size_t vb = ((size_t)(b * N_HEADSc + h)) * D_HEADc * SEQc;
  Vt[vb + (size_t)(2 * i) * SEQc + t]     = QKV[src + 2048];
  Vt[vb + (size_t)(2 * i + 1) * SEQc + t] = QKV[src + 2049];
}

// ---------------------------------------------------------------------------
// Causal flash attention.  grid = (SEQ/128, B*H), 8 waves per block,
// each wave owns 16 query rows; keys processed in chunks of 128.
// K (128x64) and V^T (64x128) chunks are staged into LDS once per block via
// async copies (ASYNCcnt), double-buffered so chunk ch+1 streams in while
// chunk ch is computed.  Online softmax in registers; P tile per-wave in LDS.
// Dynamic LDS layout (bytes):
//   Ks: 2 * 128*72*2 = 36864
//   Vs: 2 * 64*136*2 = 34816
//   Ps: 8 * 16*136*2 = 34816          total = 106496
// ---------------------------------------------------------------------------
#define FLASH_LDS_BYTES (2*128*72*2 + 2*64*136*2 + 8*16*136*2)

__global__ __launch_bounds__(256) void flash_attn_kernel(
    const __bf16* __restrict__ Qr, const __bf16* __restrict__ Kr,
    const __bf16* __restrict__ Vt, __bf16* __restrict__ attnO) {
  extern __shared__ __align__(16) char smem[];
  __bf16* Ks0 = (__bf16*)smem;                               // [2][128*72]
  __bf16* Vs0 = (__bf16*)(smem + 2 * 128 * 72 * 2);          // [2][64*136]
  __bf16* Ps  = (__bf16*)(smem + 2 * 128 * 72 * 2 + 2 * 64 * 136 * 2); // [8][16*136]

  const int bh = blockIdx.y;
  const int q0 = blockIdx.x * 128;
  const int tid = threadIdx.x, lane = tid & 31, wave = tid >> 5;
  const int lrow = lane & 15, lhi = lane >> 4;
  const int kA = lhi * 8, kB = lhi * 8;
  const int qb = q0 + wave * 16;
  const __bf16* Qb = Qr + (size_t)bh * SEQc * D_HEADc;
  const __bf16* Kb = Kr + (size_t)bh * SEQc * D_HEADc;
  const __bf16* Vb = Vt + (size_t)bh * D_HEADc * SEQc;
  __bf16* Pw = &Ps[wave * 16 * 136];

  // cooperative async staging of one 128-key chunk of K and V^T
  auto issue_chunk = [&](int buf, int kc) {
#pragma unroll
    for (int i = 0; i < 4; ++i) {           // K: 128 rows x 64 d
      int c = tid + i * 256;
      int row = c >> 3, col = (c & 7) * 8;
      async_copy_b128(lds_addr(&Ks0[buf * (128 * 72) + row * 72 + col]),
                      Kb + (size_t)(kc + row) * D_HEADc + col);
    }
#pragma unroll
    for (int i = 0; i < 4; ++i) {           // V^T: 64 d-rows x 128 keys
      int c = tid + i * 256;
      int row = c >> 4, col = (c & 15) * 8;
      async_copy_b128(lds_addr(&Vs0[buf * (64 * 136) + row * 136 + col]),
                      Vb + (size_t)row * SEQc + kc + col);
    }
  };

  v16bf aQ[2];
#pragma unroll
  for (int ks = 0; ks < 2; ++ks) {
    const __bf16* p = Qb + (size_t)(qb + lrow) * D_HEADc + ks * 32 + kA;
    aQ[ks] = cat8(*(const v8bf*)p, *(const v8bf*)(p + 16));
  }

  float mst[8], lst[8];
  v8f oacc[4];
#pragma unroll
  for (int r = 0; r < 8; ++r) { mst[r] = -INFINITY; lst[r] = 0.0f; }
#pragma unroll
  for (int j = 0; j < 4; ++j) oacc[j] = zero8();

  issue_chunk(0, 0);

  for (int ch = 0; ch <= (int)blockIdx.x; ++ch) {
    const int kc = ch * 128;
    const int buf = ch & 1;
    if (ch < (int)blockIdx.x) {
      issue_chunk(buf ^ 1, kc + 128);
      asm volatile("s_wait_asynccnt 0x8" ::: "memory");  // chunk ch landed
    } else {
      asm volatile("s_wait_asynccnt 0x0" ::: "memory");
    }
    __syncthreads();
    const __bf16* Ksb = &Ks0[buf * (128 * 72)];
    const __bf16* Vsb = &Vs0[buf * (64 * 136)];

    // S = Q * K^T  (16 q-rows x 128 keys) from LDS
    v8f sacc[8];
#pragma unroll
    for (int kt = 0; kt < 8; ++kt) sacc[kt] = zero8();
#pragma unroll
    for (int kt = 0; kt < 8; ++kt) {
      const __bf16* pk = &Ksb[(kt * 16 + lrow) * 72 + kB];
      v16bf b0 = cat8(*(const v8bf*)pk,        *(const v8bf*)(pk + 16));
      v16bf b1 = cat8(*(const v8bf*)(pk + 32), *(const v8bf*)(pk + 48));
      sacc[kt] = wmma_bf16(aQ[0], b0, sacc[kt]);
      sacc[kt] = wmma_bf16(aQ[1], b1, sacc[kt]);
    }
    // scale + causal mask + chunk row-max
    float mc[8];
#pragma unroll
    for (int r = 0; r < 8; ++r) mc[r] = -INFINITY;
#pragma unroll
    for (int kt = 0; kt < 8; ++kt)
#pragma unroll
      for (int r = 0; r < 8; ++r) {
        float s = sacc[kt][r] * 0.125f;
        int col = kc + kt * 16 + lrow;
        int rowq = qb + r + lhi * 8;
        if (col > rowq) s = -INFINITY;
        sacc[kt][r] = s;
        mc[r] = fmaxf(mc[r], s);
      }
#pragma unroll
    for (int off = 1; off < 16; off <<= 1)
#pragma unroll
      for (int r = 0; r < 8; ++r) mc[r] = fmaxf(mc[r], __shfl_xor(mc[r], off, 32));
    float sf[8];
#pragma unroll
    for (int r = 0; r < 8; ++r) {
      float mn = fmaxf(mst[r], mc[r]);
      sf[r] = __expf(mst[r] - mn);
      mst[r] = mn;
    }
    // P = exp(S - m), stage to LDS, accumulate row sums
    float rs[8];
#pragma unroll
    for (int r = 0; r < 8; ++r) rs[r] = 0.0f;
#pragma unroll
    for (int kt = 0; kt < 8; ++kt)
#pragma unroll
      for (int r = 0; r < 8; ++r) {
        float p = __expf(sacc[kt][r] - mst[r]);
        rs[r] += p;
        Pw[(r + lhi * 8) * 136 + kt * 16 + lrow] = (__bf16)p;
      }
#pragma unroll
    for (int off = 1; off < 16; off <<= 1)
#pragma unroll
      for (int r = 0; r < 8; ++r) rs[r] += __shfl_xor(rs[r], off, 32);
#pragma unroll
    for (int r = 0; r < 8; ++r) lst[r] = lst[r] * sf[r] + rs[r];
#pragma unroll
    for (int j = 0; j < 4; ++j)
#pragma unroll
      for (int r = 0; r < 8; ++r) oacc[j][r] *= sf[r];
    // intra-wave LDS store->load ordering (DS ops from one wave are in order)
    asm volatile("s_wait_dscnt 0" ::: "memory");
    // O += P * V from LDS
#pragma unroll
    for (int ks = 0; ks < 4; ++ks) {
      const __bf16* pp = &Pw[lrow * 136 + ks * 32 + kA];
      v16bf aP = cat8(*(const v8bf*)pp, *(const v8bf*)(pp + 16));
#pragma unroll
      for (int j = 0; j < 4; ++j) {
        const __bf16* pv = &Vsb[(j * 16 + lrow) * 136 + ks * 32 + kB];
        v16bf bV = cat8(*(const v8bf*)pv, *(const v8bf*)(pv + 16));
        oacc[j] = wmma_bf16(aP, bV, oacc[j]);
      }
    }
    __syncthreads();  // all waves done with buf before it is overwritten
  }

  const int b = bh / N_HEADSc, h = bh % N_HEADSc;
#pragma unroll
  for (int r = 0; r < 8; ++r) {
    float rl = __builtin_amdgcn_rcpf(lst[r]);
#pragma unroll
    for (int j = 0; j < 4; ++j) {
      int t = qb + r + lhi * 8;
      int d = j * 16 + lrow;
      attnO[((size_t)(b * SEQc + t)) * D_MODELc + h * D_HEADc + d] =
          (__bf16)(oacc[j][r] * rl);
    }
  }
}

// ---------------------------------------------------------------------------
extern "C" void kernel_launch(void* const* d_in, const int* in_sizes, int n_in,
                              void* d_out, int out_size, void* d_ws, size_t ws_size,
                              hipStream_t stream) {
  (void)in_sizes; (void)n_in; (void)out_size; (void)ws_size;
  const float* resid_pre = (const float*)d_in[0];
  const float* ln1_w = (const float*)d_in[1];
  const float* ln1_b = (const float*)d_in[2];
  const float* W_Q  = (const float*)d_in[3];
  const float* b_Q  = (const float*)d_in[4];
  const float* W_K  = (const float*)d_in[5];
  const float* b_K  = (const float*)d_in[6];
  const float* W_V  = (const float*)d_in[7];
  const float* b_V  = (const float*)d_in[8];
  const float* W_O  = (const float*)d_in[9];
  const float* b_O  = (const float*)d_in[10];
  const float* ln2_w = (const float*)d_in[11];
  const float* ln2_b = (const float*)d_in[12];
  const float* W_in  = (const float*)d_in[13];
  const float* b_in  = (const float*)d_in[14];
  const float* W_out = (const float*)d_in[15];
  const float* b_out = (const float*)d_in[16];

  char* ws = (char*)d_ws;
  const size_t MB = 1ull << 20;
  __bf16* WqT   = (__bf16*)(ws + 0 * MB);   // 2 MB; WqT|WkT|WvT = combined [3072][1024]
  __bf16* WkT   = (__bf16*)(ws + 2 * MB);   // 2 MB
  __bf16* WvT   = (__bf16*)(ws + 4 * MB);   // 2 MB
  __bf16* WoT   = (__bf16*)(ws + 6 * MB);   // 2 MB
  __bf16* WinT  = (__bf16*)(ws + 8 * MB);   // 8 MB  (4096 x 1024)
  __bf16* WoutT = (__bf16*)(ws + 16 * MB);  // 8 MB  (1024 x 4096)
  float*  resid_mid = (float*)(ws + 24 * MB); // 16 MB
  __bf16* attnO = (__bf16*)(ws + 40 * MB);  // 8 MB
  __bf16* Qr    = (__bf16*)(ws + 48 * MB);  // 8 MB
  __bf16* Kr    = (__bf16*)(ws + 56 * MB);  // 8 MB
  __bf16* Vt    = (__bf16*)(ws + 64 * MB);  // 8 MB
  __bf16* xbf   = (__bf16*)(ws + 72 * MB);  // 8 MB (also LN2 output)
  __bf16* QKVtmp = (__bf16*)(ws + 80 * MB); // 24 MB (4096 x 3072)
  __bf16* hbuf  = (__bf16*)(ws + 80 * MB);  // 32 MB, reuses dead QKV scratch
  float*  biasQKV = (float*)(ws + 112 * MB); // 12 KB packed b_Q|b_K|b_V

  // Weights -> bf16, transposed to [N][K]
  transpose_w_kernel<<<(16 * 1024 * 64) / 256, 256, 0, stream>>>(W_Q, WqT, 16, 1024, 64);
  transpose_w_kernel<<<(16 * 1024 * 64) / 256, 256, 0, stream>>>(W_K, WkT, 16, 1024, 64);
  transpose_w_kernel<<<(16 * 1024 * 64) / 256, 256, 0, stream>>>(W_V, WvT, 16, 1024, 64);
  transpose_w_kernel<<<(1024 * 1024) / 256, 256, 0, stream>>>(W_O, WoT, 1, 1024, 1024);
  transpose_w_kernel<<<(1024 * 4096) / 256, 256, 0, stream>>>(W_in, WinT, 1, 1024, 4096);
  transpose_w_kernel<<<(4096 * 1024) / 256, 256, 0, stream>>>(W_out, WoutT, 1, 4096, 1024);

  // Pack QKV biases contiguously (device-to-device, capture-safe)
  hipMemcpyAsync(biasQKV,        b_Q, 1024 * sizeof(float), hipMemcpyDeviceToDevice, stream);
  hipMemcpyAsync(biasQKV + 1024, b_K, 1024 * sizeof(float), hipMemcpyDeviceToDevice, stream);
  hipMemcpyAsync(biasQKV + 2048, b_V, 1024 * sizeof(float), hipMemcpyDeviceToDevice, stream);

  // LN1
  ln_bf16_kernel<<<NTOKc, 256, 0, stream>>>(resid_pre, ln1_w, ln1_b, xbf);

  // Fused QKV projection: [4096 x 1024] x [3072 x 1024]^T
  dim3 gqkv(3072 / 128, NTOKc / 128);
  gemm_kernel<0, 0, 1><<<gqkv, 256, 0, stream>>>(
      xbf, WqT, biasQKV, nullptr, nullptr, QKVtmp, 3072, 1024);

  // RoPE(Q,K) + V transpose into per-head layouts
  reformat_qkv_kernel<<<(BATCHc * SEQc * N_HEADSc * 32) / 256, 256, 0, stream>>>(
      QKVtmp, Qr, Kr, Vt);

  // Causal flash attention (dynamic LDS: K/V double buffers + P staging)
  dim3 ga(SEQc / 128, BATCHc * N_HEADSc);
  flash_attn_kernel<<<ga, 256, FLASH_LDS_BYTES, stream>>>(Qr, Kr, Vt, attnO);

  // attn_out projection + residual -> resid_mid (f32)
  dim3 g1(1024 / 128, NTOKc / 128);
  gemm_kernel<0, 1, 0><<<g1, 256, 0, stream>>>(
      attnO, WoT, b_O, resid_pre, resid_mid, nullptr, 1024, 1024);

  // LN2
  ln_bf16_kernel<<<NTOKc, 256, 0, stream>>>(resid_mid, ln2_w, ln2_b, xbf);

  // MLP up + GELU
  dim3 g2(4096 / 128, NTOKc / 128);
  gemm_kernel<1, 0, 1><<<g2, 256, 0, stream>>>(
      xbf, WinT, b_in, nullptr, nullptr, hbuf, 4096, 1024);

  // MLP down + residual -> final f32 output
  gemm_kernel<0, 1, 0><<<g1, 256, 0, stream>>>(
      hbuf, WoutT, b_out, resid_mid, (float*)d_out, nullptr, 1024, 4096);
}
